// DenoisingEGNN_89335319757360
// MI455X (gfx1250) — matrix operand order repaired and checked by reading, not compiled, not run on hardware
//
#include <hip/hip_runtime.h>

// ---------------------------------------------------------------------------
// EGNN denoiser forward for MI455X (gfx1250, wave32, WMMA + TDM).
// Edge/Node MLPs run on v_wmma_f32_16x16x32_bf16 with f32 accumulation.
// Weights are pre-packed into a contiguous-k A-fragment layout (k = 16h + s);
// activations are staged edge-major in LDS so B fragments are contiguous
// 32B-per-lane loads. The per-edge h-row gathers use the Tensor Data Mover
// in gather mode (16 x 16-bit row indices -> 16x128 bf16 LDS tile), waited
// with s_wait_tensorcnt. Scatters (segment_sum) use global f32 atomics.
// ---------------------------------------------------------------------------

#define NN 50000   // nodes
#define NE 800000  // edges
#define NBX 256    // batch
#define HD 128     // hidden
#define NL 4       // layers

typedef __attribute__((ext_vector_type(16))) __bf16 v16bf;
typedef __attribute__((ext_vector_type(8)))  float  v8f;
typedef __attribute__((ext_vector_type(4)))  unsigned int u32x4;
typedef __attribute__((ext_vector_type(8)))  int i32x8;
typedef __attribute__((ext_vector_type(4)))  int i32x4;

#if __has_builtin(__builtin_amdgcn_tensor_load_to_lds) && __has_builtin(__builtin_amdgcn_s_wait_tensorcnt)
#define USE_TDM_GATHER 1
#else
#define USE_TDM_GATHER 0
#endif

__device__ __forceinline__ float silu_f(float v) { return v / (1.0f + __expf(-v)); }

__device__ __forceinline__ unsigned short f2bf(float f) {
  unsigned int u = __float_as_uint(f);
  u += 0x7FFFu + ((u >> 16) & 1u);          // round-to-nearest-even
  return (unsigned short)(u >> 16);
}
__device__ __forceinline__ float bf2f(unsigned short s) {
  return __uint_as_float(((unsigned int)s) << 16);
}
__device__ __forceinline__ v16bf ldfrag(const unsigned short* p) {
  return *reinterpret_cast<const v16bf*>(p);
}
__device__ __forceinline__ v8f wmma_bf16(v16bf a, v16bf b, v8f c) {
  return __builtin_amdgcn_wmma_f32_16x16x32_bf16(false, a, false, b, (short)0, c, false, false);
}

#if USE_TDM_GATHER
// TDM gather-mode load: 16 rows (16-bit indices) x 128 bf16 from hbf -> LDS tile.
// D# built per CDNA5 ISA §8.3/§8.4/§8.7: count=1, gather_mode=1, idx16,
// data_size=2B, tensor_dim0=stride=128, tile_dim0=128, tile_dim1=16 indices.
// This toolchain exposes the 6-arg builtin: (g0 u32x4, g1 i32x8, g2 i32x4,
// g3 i32x4, g4 i32x8 [second 8-SGPR slot, unused for <=2D], cpol i32).
__device__ __forceinline__ void tdm_gather16(const unsigned short* gbase,
                                             unsigned lds_byte_off,
                                             const int* idx /*16 row ids*/) {
  unsigned long long ga = (unsigned long long)(uintptr_t)gbase;
  u32x4 g0;
  g0[0] = 0x80000001u;                       // count=1 | gather_mode=1 | idx16
  g0[1] = lds_byte_off;                      // lds_addr
  g0[2] = (unsigned)(ga & 0xFFFFFFFFull);    // global_addr[31:0]
  g0[3] = ((unsigned)(ga >> 32) & 0x01FFFFFFu) | 0x80000000u;  // addr[56:32] | type=2
  i32x8 g1;
  g1[0] = 0x00010000;                        // data_size=2B, wg_mask=0
  g1[1] = (int)(128u << 16);                 // tensor_dim0[15:0] in bits 63:48
  g1[2] = (int)(16u << 16);                  // tensor_dim0 hi=0, tensor_dim1 lo=16
  g1[3] = (int)(128u << 16);                 // tensor_dim1 hi=0, tile_dim0=128
  g1[4] = 16;                                // tile_dim1 = #valid indices
  g1[5] = 128;                               // tensor_dim0_stride lo
  g1[6] = 0;
  g1[7] = 0;
  i32x4 g2, g3;
#pragma unroll
  for (int k = 0; k < 4; ++k)
    g2[k] = (idx[2 * k] & 0xFFFF) | ((idx[2 * k + 1] & 0xFFFF) << 16);
#pragma unroll
  for (int k = 0; k < 4; ++k)
    g3[k] = (idx[8 + 2 * k] & 0xFFFF) | ((idx[8 + 2 * k + 1] & 0xFFFF) << 16);
  i32x8 g4 = (i32x8){0, 0, 0, 0, 0, 0, 0, 0};  // unused second 8-SGPR slot
  __builtin_amdgcn_tensor_load_to_lds(g0, g1, g2, g3, g4, 0);
}
#endif

// ---------------------------------------------------------------------------
// Pack W[Kin][Nout] (f32, row-major) into bf16 A-fragments for A = W^T tiles.
// Fragment (nt, kc) covers outF rows nt*16..+15 (M) and inF cols kc*32..+31 (K).
// Contiguous-k 16-bit A 16x32 layout (B fragments use the same k order):
//   lane l (l<16): M=l, K = {0..15};  lane l+16: M=l, K = {16..31}
//   element j: K = (l<16?0:16) + j
// Packed linear layout: dst[((nt*KC + kc)*32 + lane)*16 + j]
// ---------------------------------------------------------------------------
__global__ void pack_a_k(const float* __restrict__ W, unsigned short* __restrict__ dst,
                         int Kin, int Nout) {
  int idx = blockIdx.x * blockDim.x + threadIdx.x;
  if (idx >= Kin * Nout) return;
  int frag = idx >> 9;
  int r    = idx & 511;
  int lane = r >> 4;
  int j    = r & 15;
  int KC   = Kin >> 5;
  int nt   = frag / KC;
  int kc   = frag - nt * KC;
  int g    = j >> 1;
  int kpart = ((lane < 16) ? 0 : 16) + j;
  int k = kc * 32 + kpart;
  int n = nt * 16 + (lane & 15);
  dst[idx] = f2bf(W[(size_t)k * Nout + n]);
}

// ---------------------------------------------------------------------------
// Time + context conditioning: cond[b] = t_emb[b] + c_emb[b]   (B x H)
// ---------------------------------------------------------------------------
__global__ __launch_bounds__(128) void cond_k(
    const float* __restrict__ ts, const float* __restrict__ cdg, const float* __restrict__ cst,
    const float* __restrict__ tw1, const float* __restrict__ tb1,
    const float* __restrict__ tw2, const float* __restrict__ tb2,
    const float* __restrict__ cw1, const float* __restrict__ cb1,
    const float* __restrict__ cw2, const float* __restrict__ cb2,
    float* __restrict__ cond) {
  __shared__ float te[HD], t1[HD], c1[HD];
  const int b = blockIdx.x, j = threadIdx.x;
  const float T = ts[b];
  const float fr = __expf(-logf(10000.0f) * (float)(j & 63) / 63.0f);
  const float ang = T * fr;
  te[j] = (j < 64) ? __sinf(ang) : __cosf(ang);
  const float x0 = cdg[b], x1 = cst[b];
  __syncthreads();
  float s = 0.0f;
  for (int k = 0; k < HD; ++k) s += te[k] * tw1[k * HD + j];
  t1[j] = silu_f(s + tb1[j]);
  c1[j] = silu_f(x0 * cw1[j] + x1 * cw1[HD + j] + cb1[j]);
  __syncthreads();
  float s2 = 0.0f, s3 = 0.0f;
  for (int k = 0; k < HD; ++k) {
    s2 += t1[k] * tw2[k * HD + j];
    s3 += c1[k] * cw2[k * HD + j];
  }
  cond[b * HD + j] = s2 + tb2[j] + s3 + cb2[j];
}

// h = node_embed[z] + cond[batch]; write f32 master + bf16 copy
__global__ void inith_k(const int* __restrict__ z, const int* __restrict__ batch,
                        const float* __restrict__ emb, const float* __restrict__ cond,
                        float* __restrict__ hf, unsigned short* __restrict__ hbf) {
  int tid = blockIdx.x * blockDim.x + threadIdx.x;
  if (tid >= NN * HD) return;
  int n = tid >> 7, j = tid & 127;
  float v = emb[(size_t)z[n] * HD + j] + cond[(size_t)batch[n] * HD + j];
  hf[tid]  = v;
  hbf[tid] = f2bf(v);
}

// ---------------------------------------------------------------------------
// Edge kernel: one wave (32 lanes) per 16-edge tile. 3 fused WMMA stages.
//   m1 = silu(e_in @ W1 + b1), m = silu(m1 @ W2 + b2), p = silu(m @ Cw1 + cb1)
//   cw = p . cw2 ; scatter agg += m, xacc += diff*cw
// ---------------------------------------------------------------------------
__global__ __launch_bounds__(32) void egnn_edge(
    const int* __restrict__ eidx, const unsigned short* __restrict__ hbf,
    const float* __restrict__ x,
    const unsigned short* __restrict__ w1p, const float* __restrict__ w1r,
    const float* __restrict__ b1,
    const unsigned short* __restrict__ w2p, const float* __restrict__ b2,
    const unsigned short* __restrict__ cw1p, const float* __restrict__ cb1,
    const float* __restrict__ cw2,
    float* __restrict__ xacc, float* __restrict__ agg) {
  __shared__ alignas(32) unsigned short bufA[16 * HD];  // h[row] tile, later: m
  __shared__ alignas(32) unsigned short bufB[16 * HD];  // h[col] tile, later: p
  __shared__ alignas(32) unsigned short bufC[16 * HD];  // m1
  __shared__ int   ridx[16], cidx[16];
  __shared__ float rad[16], dif[48], part[32];

  const int lane = threadIdx.x;
  const int n16  = lane & 15;
  const int hi   = lane >> 4;      // 0 or 1
  const int mb   = hi * 8;         // C/D row base for this lane-half
  const int ebase = blockIdx.x * 16;

#if USE_TDM_GATHER
  // Uniform (wave-scalar) index fetch feeding the TDM gather descriptors.
  {
    int idr[16], idc[16];
#pragma unroll
    for (int k = 0; k < 16; ++k) {
      int e = ebase + k;
      idr[k] = (e < NE) ? eidx[e] : 0;
      idc[k] = (e < NE) ? eidx[NE + e] : 0;
    }
    tdm_gather16(hbf, (unsigned)(uintptr_t)(void*)&bufA[0], idr);
    tdm_gather16(hbf, (unsigned)(uintptr_t)(void*)&bufB[0], idc);
  }
#endif

  if (lane < 16) {
    int e = ebase + lane;
    int r = 0, c = 0;
    float d0 = 0.f, d1 = 0.f, d2 = 0.f, rd = 0.f;
    if (e < NE) {
      r = eidx[e];
      c = eidx[NE + e];
      d0 = x[r * 3 + 0] - x[c * 3 + 0];
      d1 = x[r * 3 + 1] - x[c * 3 + 1];
      d2 = x[r * 3 + 2] - x[c * 3 + 2];
      rd = fminf(d0 * d0 + d1 * d1 + d2 * d2, 100.0f);
    }
    ridx[lane] = r; cidx[lane] = c; rad[lane] = rd;
    dif[lane * 3 + 0] = d0; dif[lane * 3 + 1] = d1; dif[lane * 3 + 2] = d2;
  }
  __syncthreads();

#if USE_TDM_GATHER
  __builtin_amdgcn_s_wait_tensorcnt(0);   // h tiles resident in LDS
#else
  // manual gather: 16B per lane per step into LDS
#pragma unroll
  for (int it = 0; it < 8; ++it) {
    int rr = it * 2 + hi;
    const uint4* sa = reinterpret_cast<const uint4*>(hbf + (size_t)ridx[rr] * HD);
    const uint4* sb = reinterpret_cast<const uint4*>(hbf + (size_t)cidx[rr] * HD);
    *reinterpret_cast<uint4*>(&bufA[rr * HD + n16 * 8]) = sa[n16];
    *reinterpret_cast<uint4*>(&bufB[rr * HD + n16 * 8]) = sb[n16];
  }
  __syncthreads();
#endif

  const float rv = rad[n16];
  const bool  ev = (ebase + n16) < NE;
  const int   rn = ridx[n16];

  // ---- stage 1: m1 = silu(e_in @ W1 + b1)  (K=256 from h_row|h_col, +radial)
  for (int nt = 0; nt < 8; ++nt) {
    v8f acc;
    const float* bp = b1 + nt * 16 + mb;
    const float* wr = w1r + nt * 16 + mb;
#pragma unroll
    for (int r = 0; r < 8; ++r) acc[r] = bp[r] + rv * wr[r];
#pragma unroll
    for (int kc = 0; kc < 8; ++kc) {
      v16bf a = ldfrag(w1p + (size_t)(nt * 8 + kc) * 512 + lane * 16);
      const unsigned short* bs = (kc < 4) ? &bufA[n16 * HD + kc * 32 + hi * 16]
                                          : &bufB[n16 * HD + (kc - 4) * 32 + hi * 16];
      acc = wmma_bf16(a, ldfrag(bs), acc);
    }
    uint4 q;
    q.x = f2bf(silu_f(acc[0])) | ((unsigned)f2bf(silu_f(acc[1])) << 16);
    q.y = f2bf(silu_f(acc[2])) | ((unsigned)f2bf(silu_f(acc[3])) << 16);
    q.z = f2bf(silu_f(acc[4])) | ((unsigned)f2bf(silu_f(acc[5])) << 16);
    q.w = f2bf(silu_f(acc[6])) | ((unsigned)f2bf(silu_f(acc[7])) << 16);
    *reinterpret_cast<uint4*>(&bufC[n16 * HD + nt * 16 + mb]) = q;
  }
  __syncthreads();

  // ---- stage 2: m = silu(m1 @ W2 + b2); scatter agg += m; store m -> bufA
  for (int nt = 0; nt < 8; ++nt) {
    v8f acc;
    const float* bp = b2 + nt * 16 + mb;
#pragma unroll
    for (int r = 0; r < 8; ++r) acc[r] = bp[r];
#pragma unroll
    for (int kc = 0; kc < 4; ++kc) {
      v16bf a = ldfrag(w2p + (size_t)(nt * 4 + kc) * 512 + lane * 16);
      acc = wmma_bf16(a, ldfrag(&bufC[n16 * HD + kc * 32 + hi * 16]), acc);
    }
    float m[8];
#pragma unroll
    for (int r = 0; r < 8; ++r) m[r] = silu_f(acc[r]);
    if (ev) {
      float* ap = agg + (size_t)rn * HD + nt * 16 + mb;
#pragma unroll
      for (int r = 0; r < 8; ++r) atomicAdd(ap + r, m[r]);
    }
    uint4 q;
    q.x = f2bf(m[0]) | ((unsigned)f2bf(m[1]) << 16);
    q.y = f2bf(m[2]) | ((unsigned)f2bf(m[3]) << 16);
    q.z = f2bf(m[4]) | ((unsigned)f2bf(m[5]) << 16);
    q.w = f2bf(m[6]) | ((unsigned)f2bf(m[7]) << 16);
    *reinterpret_cast<uint4*>(&bufA[n16 * HD + nt * 16 + mb]) = q;
  }
  __syncthreads();

  // ---- stage 3: p = silu(m @ Cw1 + cb1) -> bufB
  for (int nt = 0; nt < 8; ++nt) {
    v8f acc;
    const float* bp = cb1 + nt * 16 + mb;
#pragma unroll
    for (int r = 0; r < 8; ++r) acc[r] = bp[r];
#pragma unroll
    for (int kc = 0; kc < 4; ++kc) {
      v16bf a = ldfrag(cw1p + (size_t)(nt * 4 + kc) * 512 + lane * 16);
      acc = wmma_bf16(a, ldfrag(&bufA[n16 * HD + kc * 32 + hi * 16]), acc);
    }
    uint4 q;
    q.x = f2bf(silu_f(acc[0])) | ((unsigned)f2bf(silu_f(acc[1])) << 16);
    q.y = f2bf(silu_f(acc[2])) | ((unsigned)f2bf(silu_f(acc[3])) << 16);
    q.z = f2bf(silu_f(acc[4])) | ((unsigned)f2bf(silu_f(acc[5])) << 16);
    q.w = f2bf(silu_f(acc[6])) | ((unsigned)f2bf(silu_f(acc[7])) << 16);
    *reinterpret_cast<uint4*>(&bufB[n16 * HD + nt * 16 + mb]) = q;
  }
  __syncthreads();

  // ---- stage 4: cw = p . cw2 (split dot across lane halves); scatter coords
  {
    float s = 0.0f;
    const unsigned short* pr = &bufB[n16 * HD + hi * 64];
    const float* cw = cw2 + hi * 64;
    for (int k = 0; k < 64; ++k) s += bf2f(pr[k]) * cw[k];
    part[lane] = s;
    __syncthreads();
    if (lane < 16 && ev) {
      float cwv = part[lane] + part[lane + 16];
      atomicAdd(&xacc[(size_t)rn * 3 + 0], dif[lane * 3 + 0] * cwv);
      atomicAdd(&xacc[(size_t)rn * 3 + 1], dif[lane * 3 + 1] * cwv);
      atomicAdd(&xacc[(size_t)rn * 3 + 2], dif[lane * 3 + 2] * cwv);
    }
  }
}

// ---------------------------------------------------------------------------
// Node kernel: h += MLP([h, agg]); x += xacc; re-zero agg/xacc for next layer.
// One wave per 16-node tile.
// ---------------------------------------------------------------------------
__global__ __launch_bounds__(32) void egnn_node(
    unsigned short* __restrict__ hbf, float* __restrict__ hf,
    float* __restrict__ agg, float* __restrict__ x, float* __restrict__ xacc,
    const unsigned short* __restrict__ w1p, const float* __restrict__ b1,
    const unsigned short* __restrict__ w2p, const float* __restrict__ b2) {
  __shared__ alignas(32) unsigned short bufIn[16 * 256];  // [h | agg] per node
  __shared__ alignas(32) unsigned short bufT[16 * HD];
  const int lane = threadIdx.x;
  const int n16  = lane & 15;
  const int hi   = lane >> 4;
  const int mb   = hi * 8;
  const int nb   = blockIdx.x * 16;

#pragma unroll
  for (int it = 0; it < 8; ++it) {
    int rr = it * 2 + hi;
    const uint4* sh = reinterpret_cast<const uint4*>(hbf + (size_t)(nb + rr) * HD);
    *reinterpret_cast<uint4*>(&bufIn[rr * 256 + n16 * 8]) = sh[n16];
  }
#pragma unroll 4
  for (int row = 0; row < 16; ++row) {
    float* ap = agg + (size_t)(nb + row) * HD + lane * 4;
    float4 v = *reinterpret_cast<const float4*>(ap);
    uint2 p;
    p.x = f2bf(v.x) | ((unsigned)f2bf(v.y) << 16);
    p.y = f2bf(v.z) | ((unsigned)f2bf(v.w) << 16);
    *reinterpret_cast<uint2*>(&bufIn[row * 256 + HD + lane * 4]) = p;
    *reinterpret_cast<float4*>(ap) = make_float4(0.f, 0.f, 0.f, 0.f);  // reset for next layer
  }
  __syncthreads();

  // stage 1: t = silu(n_in @ Nw1 + nb1)
  for (int nt = 0; nt < 8; ++nt) {
    v8f acc;
    const float* bp = b1 + nt * 16 + mb;
#pragma unroll
    for (int r = 0; r < 8; ++r) acc[r] = bp[r];
#pragma unroll
    for (int kc = 0; kc < 8; ++kc) {
      v16bf a = ldfrag(w1p + (size_t)(nt * 8 + kc) * 512 + lane * 16);
      acc = wmma_bf16(a, ldfrag(&bufIn[n16 * 256 + kc * 32 + hi * 16]), acc);
    }
    uint4 q;
    q.x = f2bf(silu_f(acc[0])) | ((unsigned)f2bf(silu_f(acc[1])) << 16);
    q.y = f2bf(silu_f(acc[2])) | ((unsigned)f2bf(silu_f(acc[3])) << 16);
    q.z = f2bf(silu_f(acc[4])) | ((unsigned)f2bf(silu_f(acc[5])) << 16);
    q.w = f2bf(silu_f(acc[6])) | ((unsigned)f2bf(silu_f(acc[7])) << 16);
    *reinterpret_cast<uint4*>(&bufT[n16 * HD + nt * 16 + mb]) = q;
  }
  __syncthreads();

  // stage 2: h += t @ Nw2 + nb2
  for (int nt = 0; nt < 8; ++nt) {
    v8f acc;
    const float* bp = b2 + nt * 16 + mb;
#pragma unroll
    for (int r = 0; r < 8; ++r) acc[r] = bp[r];
#pragma unroll
    for (int kc = 0; kc < 4; ++kc) {
      v16bf a = ldfrag(w2p + (size_t)(nt * 4 + kc) * 512 + lane * 16);
      acc = wmma_bf16(a, ldfrag(&bufT[n16 * HD + kc * 32 + hi * 16]), acc);
    }
    float* hp = hf + (size_t)(nb + n16) * HD + nt * 16 + mb;
    float nh[8];
#pragma unroll
    for (int r = 0; r < 8; ++r) { nh[r] = hp[r] + acc[r]; hp[r] = nh[r]; }
    uint4 q;
    q.x = f2bf(nh[0]) | ((unsigned)f2bf(nh[1]) << 16);
    q.y = f2bf(nh[2]) | ((unsigned)f2bf(nh[3]) << 16);
    q.z = f2bf(nh[4]) | ((unsigned)f2bf(nh[5]) << 16);
    q.w = f2bf(nh[6]) | ((unsigned)f2bf(nh[7]) << 16);
    *reinterpret_cast<uint4*>(hbf + (size_t)(nb + n16) * HD + nt * 16 + mb) = q;
  }
  // position update + reset accumulator
  if (lane < 16) {
    int node = nb + lane;
#pragma unroll
    for (int d = 0; d < 3; ++d) {
      x[(size_t)node * 3 + d] += xacc[(size_t)node * 3 + d];
      xacc[(size_t)node * 3 + d] = 0.0f;
    }
  }
}

// ---------------------------------------------------------------------------
// eps_x head: out = silu(h @ Xw1 + b1) @ Xw2 + b2   (H -> H -> 3)
// ---------------------------------------------------------------------------
__global__ __launch_bounds__(32) void head_epsx(
    const unsigned short* __restrict__ hbf,
    const unsigned short* __restrict__ w1p, const float* __restrict__ b1,
    const float* __restrict__ w2, const float* __restrict__ b2,
    float* __restrict__ out) {
  __shared__ alignas(32) unsigned short bufIn[16 * HD];
  __shared__ alignas(32) unsigned short bufT[16 * HD];
  __shared__ float pd[96];
  const int lane = threadIdx.x;
  const int n16  = lane & 15;
  const int hi   = lane >> 4;
  const int mb   = hi * 8;
  const int nb   = blockIdx.x * 16;

#pragma unroll
  for (int it = 0; it < 8; ++it) {
    int rr = it * 2 + hi;
    const uint4* sh = reinterpret_cast<const uint4*>(hbf + (size_t)(nb + rr) * HD);
    *reinterpret_cast<uint4*>(&bufIn[rr * HD + n16 * 8]) = sh[n16];
  }
  __syncthreads();

  for (int nt = 0; nt < 8; ++nt) {
    v8f acc;
    const float* bp = b1 + nt * 16 + mb;
#pragma unroll
    for (int r = 0; r < 8; ++r) acc[r] = bp[r];
#pragma unroll
    for (int kc = 0; kc < 4; ++kc) {
      v16bf a = ldfrag(w1p + (size_t)(nt * 4 + kc) * 512 + lane * 16);
      acc = wmma_bf16(a, ldfrag(&bufIn[n16 * HD + kc * 32 + hi * 16]), acc);
    }
    uint4 q;
    q.x = f2bf(silu_f(acc[0])) | ((unsigned)f2bf(silu_f(acc[1])) << 16);
    q.y = f2bf(silu_f(acc[2])) | ((unsigned)f2bf(silu_f(acc[3])) << 16);
    q.z = f2bf(silu_f(acc[4])) | ((unsigned)f2bf(silu_f(acc[5])) << 16);
    q.w = f2bf(silu_f(acc[6])) | ((unsigned)f2bf(silu_f(acc[7])) << 16);
    *reinterpret_cast<uint4*>(&bufT[n16 * HD + nt * 16 + mb]) = q;
  }
  __syncthreads();

  float s0 = 0.f, s1 = 0.f, s2 = 0.f;
  const unsigned short* pr = &bufT[n16 * HD + hi * 64];
  for (int k = 0; k < 64; ++k) {
    float v = bf2f(pr[k]);
    int kk = hi * 64 + k;
    s0 += v * w2[kk * 3 + 0];
    s1 += v * w2[kk * 3 + 1];
    s2 += v * w2[kk * 3 + 2];
  }
  pd[lane * 3 + 0] = s0; pd[lane * 3 + 1] = s1; pd[lane * 3 + 2] = s2;
  __syncthreads();
  if (lane < 16) {
    int node = nb + lane;
#pragma unroll
    for (int d = 0; d < 3; ++d)
      out[(size_t)node * 3 + d] = pd[lane * 3 + d] + pd[(lane + 16) * 3 + d] + b2[d];
  }
}

// segment-sum of h over batch -> gsum, counts
__global__ void gsum_k(const float* __restrict__ hf, const int* __restrict__ batch,
                       float* __restrict__ gsum, float* __restrict__ counts) {
  int tid = blockIdx.x * blockDim.x + threadIdx.x;
  int n = tid >> 5, j = tid & 31;
  if (n >= NN) return;
  int b = batch[n];
  const float* hp = hf + (size_t)n * HD + j * 4;
  float* gp = gsum + (size_t)b * HD + j * 4;
#pragma unroll
  for (int t = 0; t < 4; ++t) atomicAdd(gp + t, hp[t]);
  if (j == 0) atomicAdd(counts + b, 1.0f);
}

__device__ void scalar_head(const float* g, float* t, const float* w1, const float* b1,
                            const float* w2, const float* b2, float* outp, int j) {
  if (j < 64) {
    float s = 0.f;
    for (int k = 0; k < HD; ++k) s += g[k] * w1[k * 64 + j];
    t[j] = silu_f(s + b1[j]);
  }
  __syncthreads();
  if (j == 0) {
    float s = 0.f;
    for (int k = 0; k < 64; ++k) s += t[k] * w2[k];
    *outp = s + b2[0];
  }
  __syncthreads();
}

// lattice + dg/st/sy heads; one block per batch row
__global__ __launch_bounds__(128) void heads_k(
    const float* __restrict__ gsum, const float* __restrict__ counts,
    const float* __restrict__ lw1, const float* __restrict__ lb1,
    const float* __restrict__ lw2, const float* __restrict__ lb2,
    const float* __restrict__ dgw1, const float* __restrict__ dgb1,
    const float* __restrict__ dgw2, const float* __restrict__ dgb2,
    const float* __restrict__ stw1, const float* __restrict__ stb1,
    const float* __restrict__ stw2, const float* __restrict__ stb2,
    const float* __restrict__ syw1, const float* __restrict__ syb1,
    const float* __restrict__ syw2, const float* __restrict__ syb2,
    float* __restrict__ out) {
  __shared__ float g[HD], t[HD];
  const int b = blockIdx.x, j = threadIdx.x;
  float cnt = fmaxf(counts[b], 1.0f);
  g[j] = gsum[(size_t)b * HD + j] / cnt;
  __syncthreads();
  {
    float s = 0.f;
    for (int k = 0; k < HD; ++k) s += g[k] * lw1[k * HD + j];
    t[j] = silu_f(s + lb1[j]);
  }
  __syncthreads();
  if (j < 9) {
    float s = 0.f;
    for (int k = 0; k < HD; ++k) s += t[k] * lw2[k * 9 + j];
    float msk = (j == 0 || j == 1 || j == 3 || j == 4) ? 1.0f : 0.0f;
    out[(size_t)NN * 3 + (size_t)b * 9 + j] = (s + lb2[j]) * msk;
  }
  __syncthreads();
  float* base = out + (size_t)NN * 3 + (size_t)NBX * 9;
  scalar_head(g, t, dgw1, dgb1, dgw2, dgb2, base + b, j);
  scalar_head(g, t, stw1, stb1, stw2, stb2, base + NBX + b, j);
  scalar_head(g, t, syw1, syb1, syw2, syb2, base + 2 * NBX + b, j);
}

// ---------------------------------------------------------------------------
extern "C" void kernel_launch(void* const* d_in, const int* in_sizes, int n_in,
                              void* d_out, int out_size, void* d_ws, size_t ws_size,
                              hipStream_t stream) {
  (void)in_sizes; (void)n_in; (void)out_size; (void)ws_size;
  // inputs in setup_inputs() dict order; params flattened in insertion order
  const int*   z     = (const int*)d_in[0];
  const float* pos   = (const float*)d_in[1];
  const int*   eidx  = (const int*)d_in[2];
  const int*   batch = (const int*)d_in[3];
  const float* ts    = (const float*)d_in[4];
  const float* cdg   = (const float*)d_in[5];
  const float* cst   = (const float*)d_in[6];
  const float* node_embed = (const float*)d_in[7];
  const float* time_w1 = (const float*)d_in[8];  const float* time_b1 = (const float*)d_in[9];
  const float* time_w2 = (const float*)d_in[10]; const float* time_b2 = (const float*)d_in[11];
  const float* ctx_w1  = (const float*)d_in[12]; const float* ctx_b1  = (const float*)d_in[13];
  const float* ctx_w2  = (const float*)d_in[14]; const float* ctx_b2  = (const float*)d_in[15];
  const float* edge_w1 = (const float*)d_in[16]; const float* edge_b1 = (const float*)d_in[17];
  const float* edge_w2 = (const float*)d_in[18]; const float* edge_b2 = (const float*)d_in[19];
  const float* coord_w1 = (const float*)d_in[20]; const float* coord_b1 = (const float*)d_in[21];
  const float* coord_w2 = (const float*)d_in[22];
  const float* node_w1 = (const float*)d_in[23]; const float* node_b1 = (const float*)d_in[24];
  const float* node_w2 = (const float*)d_in[25]; const float* node_b2 = (const float*)d_in[26];
  const float* xh_w1 = (const float*)d_in[27]; const float* xh_b1 = (const float*)d_in[28];
  const float* xh_w2 = (const float*)d_in[29]; const float* xh_b2 = (const float*)d_in[30];
  const float* lat_w1 = (const float*)d_in[31]; const float* lat_b1 = (const float*)d_in[32];
  const float* lat_w2 = (const float*)d_in[33]; const float* lat_b2 = (const float*)d_in[34];
  const float* dg_w1 = (const float*)d_in[35]; const float* dg_b1 = (const float*)d_in[36];
  const float* dg_w2 = (const float*)d_in[37]; const float* dg_b2 = (const float*)d_in[38];
  const float* st_w1 = (const float*)d_in[39]; const float* st_b1 = (const float*)d_in[40];
  const float* st_w2 = (const float*)d_in[41]; const float* st_b2 = (const float*)d_in[42];
  const float* sy_w1 = (const float*)d_in[43]; const float* sy_b1 = (const float*)d_in[44];
  const float* sy_w2 = (const float*)d_in[45]; const float* sy_b2 = (const float*)d_in[46];

  // workspace carve (256B aligned regions)
  char* ws = (char*)d_ws;
  size_t off = 0;
  auto carve = [&](size_t bytes) -> char* {
    char* p = ws + off;
    off = (off + bytes + 255) & ~(size_t)255;
    return p;
  };
  float*          hf    = (float*)carve((size_t)NN * HD * 4);
  unsigned short* hbf   = (unsigned short*)carve((size_t)NN * HD * 2);
  float*          x     = (float*)carve((size_t)NN * 3 * 4);
  float*          xacc  = (float*)carve((size_t)NN * 3 * 4);
  float*          agg   = (float*)carve((size_t)NN * HD * 4);
  float*          cond  = (float*)carve((size_t)NBX * HD * 4);
  float*          gsum  = (float*)carve((size_t)NBX * HD * 4);
  float*          cnts  = (float*)carve((size_t)NBX * 4);
  unsigned short* ew1p  = (unsigned short*)carve((size_t)NL * 32768 * 2);
  unsigned short* ew2p  = (unsigned short*)carve((size_t)NL * 16384 * 2);
  unsigned short* cw1p  = (unsigned short*)carve((size_t)NL * 16384 * 2);
  unsigned short* nw1p  = (unsigned short*)carve((size_t)NL * 32768 * 2);
  unsigned short* nw2p  = (unsigned short*)carve((size_t)NL * 16384 * 2);
  unsigned short* xw1p  = (unsigned short*)carve((size_t)16384 * 2);

  (void)hipMemsetAsync(agg,  0, (size_t)NN * HD * 4, stream);
  (void)hipMemsetAsync(xacc, 0, (size_t)NN * 3 * 4, stream);
  (void)hipMemsetAsync(gsum, 0, (size_t)NBX * HD * 4, stream);
  (void)hipMemsetAsync(cnts, 0, (size_t)NBX * 4, stream);
  (void)hipMemcpyAsync(x, pos, (size_t)NN * 3 * 4, hipMemcpyDeviceToDevice, stream);

  // pack weights into WMMA A-fragment layout (bf16)
  for (int i = 0; i < NL; ++i) {
    pack_a_k<<<128, 256, 0, stream>>>(edge_w1 + (size_t)i * 257 * HD, ew1p + (size_t)i * 32768, 256, HD);
    pack_a_k<<<64, 256, 0, stream>>>(edge_w2 + (size_t)i * HD * HD, ew2p + (size_t)i * 16384, 128, HD);
    pack_a_k<<<64, 256, 0, stream>>>(coord_w1 + (size_t)i * HD * HD, cw1p + (size_t)i * 16384, 128, HD);
    pack_a_k<<<128, 256, 0, stream>>>(node_w1 + (size_t)i * 256 * HD, nw1p + (size_t)i * 32768, 256, HD);
    pack_a_k<<<64, 256, 0, stream>>>(node_w2 + (size_t)i * HD * HD, nw2p + (size_t)i * 16384, 128, HD);
  }
  pack_a_k<<<64, 256, 0, stream>>>(xh_w1, xw1p, 128, HD);

  cond_k<<<NBX, 128, 0, stream>>>(ts, cdg, cst, time_w1, time_b1, time_w2, time_b2,
                                  ctx_w1, ctx_b1, ctx_w2, ctx_b2, cond);
  inith_k<<<(NN * HD + 255) / 256, 256, 0, stream>>>(z, batch, node_embed, cond, hf, hbf);

  for (int i = 0; i < NL; ++i) {
    egnn_edge<<<NE / 16, 32, 0, stream>>>(
        eidx, hbf, x,
        ew1p + (size_t)i * 32768, edge_w1 + ((size_t)i * 257 + 256) * HD, edge_b1 + (size_t)i * HD,
        ew2p + (size_t)i * 16384, edge_b2 + (size_t)i * HD,
        cw1p + (size_t)i * 16384, coord_b1 + (size_t)i * HD,
        coord_w2 + (size_t)i * HD,
        xacc, agg);
    egnn_node<<<NN / 16, 32, 0, stream>>>(
        hbf, hf, agg, x, xacc,
        nw1p + (size_t)i * 32768, node_b1 + (size_t)i * HD,
        nw2p + (size_t)i * 16384, node_b2 + (size_t)i * HD);
  }

  head_epsx<<<NN / 16, 32, 0, stream>>>(hbf, xw1p, xh_b1, xh_w2, xh_b2, (float*)d_out);
  gsum_k<<<(NN * 32 + 255) / 256, 256, 0, stream>>>(hf, batch, gsum, cnts);
  heads_k<<<NBX, 128, 0, stream>>>(gsum, cnts, lat_w1, lat_b1, lat_w2, lat_b2,
                                   dg_w1, dg_b1, dg_w2, dg_b2,
                                   st_w1, st_b1, st_w2, st_b2,
                                   sy_w1, sy_b1, sy_w2, sy_b2,
                                   (float*)d_out);
}
